// QPUnrolledNetwork_43997644980585
// MI455X (gfx1250) — matrix operand order repaired and checked by reading, not compiled, not run on hardware
//
#include <hip/hip_runtime.h>
#include <hip/hip_bf16.h>

typedef __attribute__((ext_vector_type(16))) _Float16 v16h;
typedef __attribute__((ext_vector_type(8)))  float    v8f;
typedef unsigned int u32x4 __attribute__((ext_vector_type(4)));
typedef int          i32x4 __attribute__((ext_vector_type(4)));
typedef int          i32x8 __attribute__((ext_vector_type(8)));

#define B_ROWS 16384
#define IN_DIM 128
#define HID_DIM 512
#define N_DIM 16
#define M_DIM 32
#define NP_DIM 136           // N*(N+1)/2
#define PH_DIM 648           // NP + M*N
#define QB_DIM 48            // N + M
#define QP_ITERS 100
#define ALPHA_C 0.1f
#define MIN_EIG_C 0.01f

#if __has_builtin(__builtin_amdgcn_tensor_load_to_lds) && __has_builtin(__builtin_amdgcn_s_wait_tensorcnt)
#define USE_TDM 1
#else
#define USE_TDM 0
#endif

// ---------------------------------------------------------------------------
// f32 -> f16 conversion
// ---------------------------------------------------------------------------
__global__ void cvt_f32_f16(const float* __restrict__ in, _Float16* __restrict__ out, int n) {
    int i = blockIdx.x * 256 + threadIdx.x;
    if (i < n) out[i] = (_Float16)in[i];
}

#if USE_TDM
// ---------------------------------------------------------------------------
// Tensor Data Mover: async 2D tile load (rows x 32 halfs) global -> LDS.
// D# per CDNA5 ISA:
//  group0: [1:0]=count=1, [63:32]=lds_addr, [120:64]=global_addr, [127:126]=2
//  group1: data_size=2B, pad_enable (1 dword pad per 16 dwords -> 17-dword LDS
//          rows, matching the bank-conflict-free fragment layout),
//          tensor_dim0=K, tensor_dim1=rows_remaining (OOB rows read as zero),
//          tile = 32 x 128, tensor_dim0_stride = K.
// ---------------------------------------------------------------------------
__device__ __forceinline__ void tdm_load_tile(unsigned lds_off, const _Float16* gaddr,
                                              unsigned K_elems, unsigned rows_remain) {
    unsigned long long ga = (unsigned long long)(size_t)gaddr;
    u32x4 g0;
    g0[0] = 1u;                                                // count=1 (valid, user mode)
    g0[1] = lds_off;                                           // lds_addr (bytes)
    g0[2] = (unsigned)(ga & 0xFFFFFFFFu);                      // global_addr[31:0]
    g0[3] = (unsigned)((ga >> 32) & 0x1FFFFFFu) | (2u << 30);  // global_addr[56:32] | type=2
    i32x8 g1;
    g1[0] = (int)((1u << 16) | (1u << 20) | (3u << 22));       // data_size=2B, pad_en, ivl=16dw, amt=1dw
    g1[1] = (int)((K_elems & 0xFFFFu) << 16);                  // tensor_dim0[15:0]
    g1[2] = (int)(((K_elems >> 16) & 0xFFFFu) | ((rows_remain & 0xFFFFu) << 16)); // dim0 hi | dim1 lo
    g1[3] = (int)(((rows_remain >> 16) & 0xFFFFu) | (32u << 16)); // dim1 hi | tile_dim0=32
    g1[4] = 128;                                               // tile_dim1=128 rows, tile_dim2=0
    g1[5] = (int)K_elems;                                      // tensor_dim0_stride lo
    g1[6] = 0;                                                 // stride0 hi | stride1 lo
    g1[7] = 0;
    i32x4 z4 = {0, 0, 0, 0};
#if defined(__clang_major__) && __clang_major__ >= 23
    i32x8 z8 = {0, 0, 0, 0, 0, 0, 0, 0};
    __builtin_amdgcn_tensor_load_to_lds(g0, g1, z4, z4, z8, 0);
#else
    __builtin_amdgcn_tensor_load_to_lds(g0, g1, z4, z4, 0);
#endif
}
#endif

// ---------------------------------------------------------------------------
// Tiled WMMA GEMM:  C[row,col] = act( sum_k A[row,k] * W[col,k] + bias[col] )
// A: Brows x K (f16, row-major), W: Nout x K (f16, row-major)
// Workgroup: 128 threads = 4 waves; tile 128(M) x 128(N); each wave owns 64x64
// (4x4 grid of 16x16x32 WMMA tiles). K staged in chunks of 32 through LDS with
// double-buffered TDM async copies (TENSORcnt) overlapping WMMA compute.
// ---------------------------------------------------------------------------
template<bool RELU, bool OUT_H, bool OUT_F>
__global__ __launch_bounds__(128) void gemm_wmma(
        const _Float16* __restrict__ A, const _Float16* __restrict__ W,
        const float* __restrict__ bias,
        _Float16* __restrict__ outH, float* __restrict__ outF,
        int K, int Nout)
{
    // rows of 32 halfs = 16 dwords, padded to 17 dwords (TDM pad feature).
    __shared__ unsigned int Alds[2][128 * 17];
    __shared__ unsigned int Wlds[2][128 * 17];

    const int tid  = threadIdx.x;
    const int wid  = tid >> 5;
    const int lane = tid & 31;

    const int bm0 = blockIdx.y * 128;   // global row base of workgroup tile
    const int bn0 = blockIdx.x * 128;   // global col base of workgroup tile
    const int wrb = (wid >> 1) * 64;    // wave row base (local)
    const int wcb = (wid & 1) * 64;     // wave col base (local)

    v8f acc[4][4];
    #pragma unroll
    for (int mt = 0; mt < 4; ++mt)
        #pragma unroll
        for (int nt = 0; nt < 4; ++nt)
            acc[mt][nt] = (v8f){0.f,0.f,0.f,0.f,0.f,0.f,0.f,0.f};

    union Frag { unsigned int u[8]; v16h h; };

    const int nK   = K >> 5;            // k-chunks of 32
    const int il   = lane & 15;
    const int kbdA = (lane >> 4) * 4;   // A frag dword offset (lanes 16-31: K+8)
    const int kbdB = (lane >> 4) * 4;   // B frag dword offset (lanes 16-31: K+8)

#if USE_TDM
    const unsigned rowsA = (unsigned)((gridDim.y - blockIdx.y) * 128); // never OOB
    const unsigned rowsW = (unsigned)(Nout - bn0);                    // OOB rows -> 0
    if (wid == 0) {
        tdm_load_tile((unsigned)(size_t)&Alds[0][0], A + (size_t)bm0 * K, (unsigned)K, rowsA);
        tdm_load_tile((unsigned)(size_t)&Wlds[0][0], W + (size_t)bn0 * K, (unsigned)K, rowsW);
    }
#else
    const unsigned int* Ag = (const unsigned int*)A;   // K/2 dwords per row
    const unsigned int* Wg = (const unsigned int*)W;
    const int kd = K >> 1;
#endif

    for (int kt = 0; kt < nK; ++kt) {
        const int cur = kt & 1;
#if USE_TDM
        if (wid == 0) __builtin_amdgcn_s_wait_tensorcnt(0);
        __syncthreads();                       // stage kt visible to all waves
        if (wid == 0 && kt + 1 < nK) {         // async-prefetch stage kt+1
            tdm_load_tile((unsigned)(size_t)&Alds[cur ^ 1][0],
                          A + (size_t)bm0 * K + (size_t)(kt + 1) * 32, (unsigned)K, rowsA);
            tdm_load_tile((unsigned)(size_t)&Wlds[cur ^ 1][0],
                          W + (size_t)bn0 * K + (size_t)(kt + 1) * 32, (unsigned)K, rowsW);
        }
#else
        for (int idx = tid; idx < 2048; idx += 128) {
            int r = idx >> 4, c = idx & 15;
            Alds[cur][r * 17 + c] = Ag[(size_t)(bm0 + r) * kd + kt * 16 + c];
            int gr = bn0 + r;
            Wlds[cur][r * 17 + c] = (gr < Nout) ? Wg[(size_t)gr * kd + kt * 16 + c] : 0u;
        }
        __syncthreads();
#endif

        // ---- load fragments (documented CDNA5 16-bit A/B layouts) ----
        Frag af[4], bf[4];
        #pragma unroll
        for (int mt = 0; mt < 4; ++mt) {
            int base = (wrb + mt * 16 + il) * 17;
            #pragma unroll
            for (int p = 0; p < 8; ++p) {
                int col = (p < 4 ? p : p + 4) + kbdA;  // K pairs 0..7 / 16..23 (+8 hi lanes)
                af[mt].u[p] = Alds[cur][base + col];
            }
        }
        #pragma unroll
        for (int nt = 0; nt < 4; ++nt) {
            int base = (wcb + nt * 16 + il) * 17;      // B[k][n] = W[n][k]
            #pragma unroll
            for (int p = 0; p < 8; ++p)
                bf[nt].u[p] = Wlds[cur][base + (p < 4 ? p : p + 4) + kbdB];
        }

        // ---- 16 WMMAs ----
        #pragma unroll
        for (int mt = 0; mt < 4; ++mt)
            #pragma unroll
            for (int nt = 0; nt < 4; ++nt)
                acc[mt][nt] = __builtin_amdgcn_wmma_f32_16x16x32_f16(
                    false, af[mt].h, false, bf[nt].h,
                    (short)0, acc[mt][nt], false, false);

#if !USE_TDM
        __syncthreads();
#endif
    }

    // ---- epilogue: bias, relu, store (C/D: M = p + 8*(lane>=16), N = lane&15) ----
    const int mb = (lane >> 4) * 8;
    #pragma unroll
    for (int nt = 0; nt < 4; ++nt) {
        int col = bn0 + wcb + nt * 16 + il;
        float bval = (col < Nout) ? bias[col] : 0.f;
        #pragma unroll
        for (int mt = 0; mt < 4; ++mt) {
            #pragma unroll
            for (int p = 0; p < 8; ++p) {
                float v = acc[mt][nt][p] + bval;
                if (RELU) v = fmaxf(v, 0.f);
                int row = bm0 + wrb + mt * 16 + p + mb;
                if (col < Nout) {
                    if (OUT_F) outF[(size_t)row * Nout + col] = v;
                    if (OUT_H) outH[(size_t)row * Nout + col] = (_Float16)v;
                }
            }
        }
    }
}

// ---------------------------------------------------------------------------
// Fused PSD-build + QP iteration kernel. One wave32 per batch item:
//   lane m holds mu_m (M = 32 = wave width). Pinv (swizzled), H[m][i], Ht[i][m]
//   live in LDS; cross-lane reductions via __shfl. No barriers in the loop.
// ---------------------------------------------------------------------------
__global__ __launch_bounds__(256) void qp_solve(
        const float* __restrict__ ph, const float* __restrict__ qb,
        float* __restrict__ out)
{
    __shared__ float tri_s[8][NP_DIM];
    __shared__ float P_s[8][256];       // swizzled: P[i][j] at [i*16 + ((j+i)&15)]
    __shared__ float H_s[8][512];       // H[m][i]  at [m*16+i]
    __shared__ float Ht_s[8][512];      // H[m][i]  at [i*32+m]

    const int wid  = threadIdx.x >> 5;
    const int lane = threadIdx.x & 31;
    const int bb   = blockIdx.x * 8 + wid;
    const int il   = lane & 15;

    const float* phb = ph + (size_t)bb * PH_DIM;

    // stage Cholesky params + H
    for (int t = lane; t < NP_DIM; t += 32) tri_s[wid][t] = phb[t];
    for (int t = lane; t < 512; t += 32) {
        float v = phb[NP_DIM + t];
        int m = t >> 4, i = t & 15;
        H_s[wid][m * 16 + i]  = v;
        Ht_s[wid][i * 32 + m] = v;
    }
    float ql = qb[(size_t)bb * QB_DIM + il];          // q_{lane&15}
    float bl = qb[(size_t)bb * QB_DIM + 16 + lane];   // b_lane
    __syncthreads();

    // Pinv = L L^T + MIN_EIG*I  (8 entries per lane), stored swizzled
    for (int e = lane; e < 256; e += 32) {
        int i = e >> 4, j = e & 15;
        int kmax = i < j ? i : j;
        float s = (i == j) ? MIN_EIG_C : 0.f;
        const float* Li = &tri_s[wid][i * (i + 1) / 2];
        const float* Lj = &tri_s[wid][j * (j + 1) / 2];
        for (int k = 0; k <= kmax; ++k) s += Li[k] * Lj[k];
        P_s[wid][i * 16 + ((j + i) & 15)] = s;
    }
    __syncthreads();

    const float* Hw  = H_s[wid];
    const float* Htw = Ht_s[wid];
    const float* Pw  = P_s[wid];

    float mu = 0.f;
    float z  = 0.f;

    #pragma unroll 1
    for (int it = 0; it <= QP_ITERS; ++it) {
        // t_i = (H^T mu)_i - q_i   (replicated on lanes i and i+16)
        float v = -ql;
        #pragma unroll
        for (int m = 0; m < 32; ++m)
            v = fmaf(Hw[m * 16 + il], __shfl(mu, m, 32), v);
        // z_i = sum_j Pinv[i][j] * t_j
        z = 0.f;
        #pragma unroll
        for (int j = 0; j < 16; ++j)
            z = fmaf(Pw[il * 16 + ((j + il) & 15)], __shfl(v, j, 32), z);
        if (it == QP_ITERS) break;
        // w_m = (H z)_m + b_m ;  mu = relu(mu - alpha*w)
        float w = bl;
        #pragma unroll
        for (int i2 = 0; i2 < 16; ++i2)
            w = fmaf(Htw[i2 * 32 + lane], __shfl(z, i2, 32), w);
        mu = fmaxf(fmaf(-ALPHA_C, w, mu), 0.f);
    }

    if (lane < 16) out[(size_t)bb * N_DIM + lane] = z;
}

// ---------------------------------------------------------------------------
// Host-side launcher
// ---------------------------------------------------------------------------
extern "C" void kernel_launch(void* const* d_in, const int* in_sizes, int n_in,
                              void* d_out, int out_size, void* d_ws, size_t ws_size,
                              hipStream_t stream) {
    const float* x   = (const float*)d_in[0];
    const float* W1  = (const float*)d_in[1];
    const float* b1  = (const float*)d_in[2];
    const float* W2  = (const float*)d_in[3];
    const float* b2  = (const float*)d_in[4];
    const float* W3  = (const float*)d_in[5];
    const float* b3  = (const float*)d_in[6];
    const float* Wqb = (const float*)d_in[7];
    const float* bqb = (const float*)d_in[8];
    float* out = (float*)d_out;

    // workspace carve-up (256B aligned)
    char* ws = (char*)d_ws;
    size_t off = 0;
    auto carve = [&](size_t bytes) -> char* {
        char* p = ws + off;
        off = (off + bytes + 255) & ~(size_t)255;
        return p;
    };
    _Float16* x16  = (_Float16*)carve((size_t)B_ROWS * IN_DIM  * 2);
    _Float16* W1h  = (_Float16*)carve((size_t)HID_DIM * IN_DIM * 2);
    _Float16* W2h  = (_Float16*)carve((size_t)HID_DIM * HID_DIM * 2);
    _Float16* W3h  = (_Float16*)carve((size_t)PH_DIM * HID_DIM * 2);
    _Float16* Wqbh = (_Float16*)carve((size_t)QB_DIM * IN_DIM * 2);
    _Float16* h1h  = (_Float16*)carve((size_t)B_ROWS * HID_DIM * 2);
    _Float16* h2h  = (_Float16*)carve((size_t)B_ROWS * HID_DIM * 2);
    float*    phB  = (float*)carve((size_t)B_ROWS * PH_DIM * 4);
    float*    qbB  = (float*)carve((size_t)B_ROWS * QB_DIM * 4);
    (void)ws_size; (void)n_in; (void)in_sizes; (void)out_size;

    auto cvt = [&](const float* src, _Float16* dst, int n) {
        cvt_f32_f16<<<(n + 255) / 256, 256, 0, stream>>>(src, dst, n);
    };
    cvt(x,   x16,  B_ROWS * IN_DIM);
    cvt(W1,  W1h,  HID_DIM * IN_DIM);
    cvt(W2,  W2h,  HID_DIM * HID_DIM);
    cvt(W3,  W3h,  PH_DIM * HID_DIM);
    cvt(Wqb, Wqbh, QB_DIM * IN_DIM);

    const int mTiles = B_ROWS / 128;
    // h1 = relu(x @ W1^T + b1)
    gemm_wmma<true, true, false><<<dim3(HID_DIM / 128, mTiles), 128, 0, stream>>>(
        x16, W1h, b1, h1h, nullptr, IN_DIM, HID_DIM);
    // h2 = relu(h1 @ W2^T + b2)
    gemm_wmma<true, true, false><<<dim3(HID_DIM / 128, mTiles), 128, 0, stream>>>(
        h1h, W2h, b2, h2h, nullptr, HID_DIM, HID_DIM);
    // ph = h2 @ W3^T + b3   (Nout = 648, 6 col tiles; TDM zero-fills OOB rows)
    gemm_wmma<false, false, true><<<dim3((PH_DIM + 127) / 128, mTiles), 128, 0, stream>>>(
        h2h, W3h, b3, nullptr, phB, HID_DIM, PH_DIM);
    // qb = x @ Wqb^T + bqb  (Nout = 48)
    gemm_wmma<false, false, true><<<dim3(1, mTiles), 128, 0, stream>>>(
        x16, Wqbh, bqb, nullptr, qbB, IN_DIM, QB_DIM);

    // fused PSD build + 100 QP iterations + final primal -> out (B x 16)
    qp_solve<<<B_ROWS / 8, 256, 0, stream>>>(phB, qbB, out);
}
